// PagedLTSMemory_34677565948789
// MI455X (gfx1250) — matrix operand order repaired and checked by reading, not compiled, not run on hardware
//
#include <hip/hip_runtime.h>

// Problem constants (fixed by the reference).
#define BATCH   2
#define TSEQ    512
#define DMEM    128
#define NROWS   131072      // N_PAGES * PAGE_SIZE
#define TOPK    64
#define QTILE   16          // queries per workgroup (one WMMA M-tile)
#define CHUNK   512         // memory rows scored per chunk
#define NWAVES  8           // 256 threads / wave32
#define NSPLIT  8           // N-dimension split for occupancy
#define NQ      (BATCH * TSEQ)

typedef __attribute__((ext_vector_type(16))) __bf16 v16bf;
typedef __attribute__((ext_vector_type(8)))  __bf16 v8bf;
typedef __attribute__((ext_vector_type(8)))  float  v8f;
typedef __attribute__((ext_vector_type(4)))  int    v4i;

typedef __attribute__((address_space(1))) v4i* gv4i_p;   // global v4i*
typedef __attribute__((address_space(3))) v4i* lv4i_p;   // LDS v4i*

#if defined(__has_builtin)
#if __has_builtin(__builtin_amdgcn_global_load_async_to_lds_b128)
#define ASYNC_OK 1
#endif
#endif
#ifndef ASYNC_OK
#define ASYNC_OK 0
#endif

static __device__ __forceinline__ void copy16_async(const float* src, float* dst) {
#if ASYNC_OK
    __builtin_amdgcn_global_load_async_to_lds_b128((gv4i_p)(src), (lv4i_p)(dst), 0, 0);
#else
    *(float4*)dst = *(const float4*)src;
#endif
}

static __device__ __forceinline__ void async_wait_all() {
#if ASYNC_OK
#if __has_builtin(__builtin_amdgcn_s_wait_asynccnt)
    __builtin_amdgcn_s_wait_asynccnt(0);
#else
    asm volatile("s_wait_asynccnt 0x0" ::: "memory");
#endif
#endif
}

static __device__ __forceinline__ unsigned ballot32(bool p) {
#if defined(__has_builtin) && __has_builtin(__builtin_amdgcn_ballot_w32)
    return __builtin_amdgcn_ballot_w32(p);
#else
    return (unsigned)__ballot(p);
#endif
}

// ---- wave32 helpers for the distributed top-64 list (2 entries per lane) ----
struct MinPos { float v; int c; };   // c = slot*32 + lane, uniform across wave after reduce

static __device__ __forceinline__ MinPos wave_min2(float v0, float v1, int lane) {
    MinPos r;
    if (v1 < v0) { r.v = v1; r.c = 32 + lane; } else { r.v = v0; r.c = lane; }
    #pragma unroll
    for (int off = 16; off; off >>= 1) {
        float ov = __shfl_xor(r.v, off, 32);
        int   oc = __shfl_xor(r.c, off, 32);
        if (ov < r.v) { r.v = ov; r.c = oc; }
    }
    return r;
}

// Stream one candidate batch (one value per lane) into the running top-64.
// Threshold (mp) is maintained incrementally: recomputed only after a real insert.
static __device__ __forceinline__ void topk_insert32(float s, int n, int lane,
                                                     float& t0, float& t1,
                                                     int& i0, int& i1, MinPos& mp) {
    unsigned mask = ballot32(s > mp.v);
    while (mask) {
        const int src = __builtin_ctz(mask);
        mask &= mask - 1;
        const float sc = __shfl(s, src, 32);
        const int   nn = __shfl(n, src, 32);
        if (sc > mp.v) {
            if (lane == (mp.c & 31)) {
                if (mp.c & 32) { t1 = sc; i1 = nn; }
                else           { t0 = sc; i0 = nn; }
            }
            mp = wave_min2(t0, t1, lane);
        }
    }
}

// Softmax over the wave-distributed top-64 and weighted gather-reduce to out[q].
static __device__ __forceinline__ void softmax_gather_out(int q, int lane,
                                                          float t0, float t1, int i0, int i1,
                                                          const float* __restrict__ pagesf,
                                                          float* __restrict__ out) {
    const float scale = 0.08838834764831845f;   // 1/sqrt(128)
    float x0 = t0 * scale, x1 = t1 * scale;
    float mx = fmaxf(x0, x1);
    #pragma unroll
    for (int off = 16; off; off >>= 1) mx = fmaxf(mx, __shfl_xor(mx, off, 32));
    const float e0 = __expf(x0 - mx), e1 = __expf(x1 - mx);
    float sum = e0 + e1;
    #pragma unroll
    for (int off = 16; off; off >>= 1) sum += __shfl_xor(sum, off, 32);
    const float a0 = e0 / sum, a1 = e1 / sum;

    const int dbase = lane * 4;
    float acc0 = 0.f, acc1 = 0.f, acc2 = 0.f, acc3 = 0.f;
    for (int k = 0; k < TOPK; ++k) {
        const int srcl = k & 31, slot = k >> 5;
        const float av  = __shfl(slot ? a1 : a0, srcl, 32);
        const int   idx = __shfl(slot ? i1 : i0, srcl, 32);
        const float4 mv4 = *(const float4*)(pagesf + (size_t)idx * DMEM + dbase);
        acc0 = fmaf(av, mv4.x, acc0);
        acc1 = fmaf(av, mv4.y, acc1);
        acc2 = fmaf(av, mv4.z, acc2);
        acc3 = fmaf(av, mv4.w, acc3);
    }
    float4* op = (float4*)(out + (size_t)q * DMEM + dbase);
    *op = make_float4(acc0, acc1, acc2, acc3);
}

// One-time fp32 -> bf16 conversion of the page pool into workspace.
__global__ __launch_bounds__(256)
void convert_pages_bf16(const float* __restrict__ in, __bf16* __restrict__ outb) {
    const size_t i = ((size_t)blockIdx.x * 256 + threadIdx.x) * 8;
    const float4 a = *(const float4*)(in + i);
    const float4 b = *(const float4*)(in + i + 4);
    v8bf o;
    o[0] = (__bf16)a.x; o[1] = (__bf16)a.y; o[2] = (__bf16)a.z; o[3] = (__bf16)a.w;
    o[4] = (__bf16)b.x; o[5] = (__bf16)b.y; o[6] = (__bf16)b.z; o[7] = (__bf16)b.w;
    *(v8bf*)(outb + i) = o;
}

// Main scorer. SPLIT=true: bf16 pool, grid (64, NSPLIT), emits per-split top-64
// candidates. SPLIT=false: fallback, fp32 + inline cvt, full N, direct output.
template<bool SPLIT>
__global__ __launch_bounds__(256)
void paged_score_kernel(const float*  __restrict__ query,
                        const float*  __restrict__ pagesf,   // [NROWS, DMEM] fp32
                        const __bf16* __restrict__ pagesb,   // [NROWS, DMEM] bf16 (SPLIT)
                        float* __restrict__ candS,           // [NQ, NSPLIT, 64]
                        int*   __restrict__ candI,           // [NQ, NSPLIT, 64]
                        float* __restrict__ out) {
    __shared__ float qf[QTILE * DMEM];        // fp32 query tile staged via async, 8 KB
    __shared__ float scl[QTILE * CHUNK];      // chunk scores, 32 KB

    const int tid   = threadIdx.x;
    const int lane  = tid & 31;
    const int wave  = tid >> 5;
    const int half  = lane >> 4;              // 0: lanes 0-15, 1: lanes 16-31
    const int m     = lane & 15;              // row-in-tile / column-in-tile id
    const int qbase = blockIdx.x * QTILE;     // flattened (b*t) query row base
    const int split = SPLIT ? blockIdx.y : 0;
    const int rowsPerSplit = SPLIT ? (NROWS / NSPLIT) : NROWS;
    const int rowStart = split * rowsPerSplit;

    // ---- Stage the 16x128 fp32 query tile into LDS (async on gfx1250) ----
    {
        const float* gq = query + (size_t)qbase * DMEM;
        for (int s = tid; s < (QTILE * DMEM) / 4; s += 256)
            copy16_async(gq + s * 4, qf + s * 4);
        async_wait_all();
    }
    __syncthreads();

    // ---- Build the A matrix (16x32 bf16 per K-step) in registers, ONCE ----
    v16bf areg[DMEM / 32];
    #pragma unroll
    for (int ks = 0; ks < DMEM / 32; ++ks) {
        const int kb = ks * 32;
        #pragma unroll
        for (int e = 0; e < 8; ++e)
            areg[ks][e] = (__bf16)qf[m * DMEM + kb + half * 16 + e];
        #pragma unroll
        for (int e = 0; e < 8; ++e)
            areg[ks][8 + e] = (__bf16)qf[m * DMEM + kb + half * 16 + 8 + e];
    }

    // ---- Running top-64 per owned query (each wave owns 2 of the 16 queries) ----
    float ts0q0 = -INFINITY, ts1q0 = -INFINITY, ts0q1 = -INFINITY, ts1q1 = -INFINITY;
    int   ti0q0 = 0, ti1q0 = 0, ti0q1 = 0, ti1q1 = 0;
    MinPos mp0 = {-INFINITY, lane}, mp1 = {-INFINITY, lane};

    const int nchunks = rowsPerSplit / CHUNK;
    for (int chunk = 0; chunk < nchunks; ++chunk) {
        const int rowBase0 = rowStart + chunk * CHUNK;

        // Prefetch next chunk (gfx1250 global_prefetch_b8).
        if (chunk + 1 < nchunks) {
            if (SPLIT) {
                const __bf16* pf = pagesb + (size_t)(rowBase0 + CHUNK) * DMEM;
                for (int r = tid; r < CHUNK; r += 256)
                    __builtin_prefetch(pf + (size_t)r * DMEM, 0, 0);
            } else {
                const float* pf = pagesf + (size_t)(rowBase0 + CHUNK) * DMEM;
                for (int r = tid; r < CHUNK; r += 256)
                    __builtin_prefetch(pf + (size_t)r * DMEM, 0, 0);
            }
        }

        // ---- Score: C(16q x 16rows) = A(16x128) x B(128x16), bf16 WMMA ----
        #pragma unroll
        for (int tt = 0; tt < CHUNK / (NWAVES * 16); ++tt) {
            const int colBase = wave * (CHUNK / NWAVES) + tt * 16;
            const int row     = rowBase0 + colBase + m;      // this lane's memory row

            v8f c = {};
            if (SPLIT) {
                const __bf16* brow = pagesb + (size_t)row * DMEM + half * 16;
                #pragma unroll
                for (int ks = 0; ks < DMEM / 32; ++ks) {
                    const v16bf b = *(const v16bf*)(brow + ks * 32);
                    c = __builtin_amdgcn_wmma_f32_16x16x32_bf16(
                            false, areg[ks], false, b, (short)0, c, false, false);
                }
            } else {
                const float* frow = pagesf + (size_t)row * DMEM + half * 16;
                #pragma unroll
                for (int ks = 0; ks < DMEM / 32; ++ks) {
                    const float4 f0 = *(const float4*)(frow + ks * 32);
                    const float4 f1 = *(const float4*)(frow + ks * 32 + 4);
                    const float4 f2 = *(const float4*)(frow + ks * 32 + 8);
                    const float4 f3 = *(const float4*)(frow + ks * 32 + 12);
                    v16bf b;
                    b[0]  = (__bf16)f0.x; b[1]  = (__bf16)f0.y; b[2]  = (__bf16)f0.z; b[3]  = (__bf16)f0.w;
                    b[4]  = (__bf16)f1.x; b[5]  = (__bf16)f1.y; b[6]  = (__bf16)f1.z; b[7]  = (__bf16)f1.w;
                    b[8]  = (__bf16)f2.x; b[9]  = (__bf16)f2.y; b[10] = (__bf16)f2.z; b[11] = (__bf16)f2.w;
                    b[12] = (__bf16)f3.x; b[13] = (__bf16)f3.y; b[14] = (__bf16)f3.z; b[15] = (__bf16)f3.w;
                    c = __builtin_amdgcn_wmma_f32_16x16x32_bf16(
                            false, areg[ks], false, b, (short)0, c, false, false);
                }
            }
            // C layout: VGPR r -> query (r + 8*half), lane m -> column (memory row).
            #pragma unroll
            for (int r = 0; r < 8; ++r)
                scl[(r + 8 * half) * CHUNK + colBase + m] = c[r];
        }
        __syncthreads();

        // ---- Merge chunk scores into per-query register top-64 ----
        {
            const int q0 = wave * 2;
            for (int i = lane; i < CHUNK; i += 32) {
                topk_insert32(scl[q0 * CHUNK + i], rowBase0 + i, lane,
                              ts0q0, ts1q0, ti0q0, ti1q0, mp0);
            }
            for (int i = lane; i < CHUNK; i += 32) {
                topk_insert32(scl[(q0 + 1) * CHUNK + i], rowBase0 + i, lane,
                              ts0q1, ts1q1, ti0q1, ti1q1, mp1);
            }
        }
        __syncthreads();   // protect scl for next chunk
    }

    const int q0 = qbase + wave * 2;
    if (SPLIT) {
        // Emit per-split candidates: [q][split][64], slot-major within the wave.
        {
            const size_t base = ((size_t)q0 * NSPLIT + split) * TOPK;
            candS[base + lane]      = ts0q0;  candI[base + lane]      = ti0q0;
            candS[base + 32 + lane] = ts1q0;  candI[base + 32 + lane] = ti1q0;
        }
        {
            const size_t base = ((size_t)(q0 + 1) * NSPLIT + split) * TOPK;
            candS[base + lane]      = ts0q1;  candI[base + lane]      = ti0q1;
            candS[base + 32 + lane] = ts1q1;  candI[base + 32 + lane] = ti1q1;
        }
    } else {
        softmax_gather_out(q0,     lane, ts0q0, ts1q0, ti0q0, ti1q0, pagesf, out);
        softmax_gather_out(q0 + 1, lane, ts0q1, ts1q1, ti0q1, ti1q1, pagesf, out);
    }
}

// Merge NSPLIT*64 candidates per query -> top-64, softmax, gather, output.
// One wave per query: grid = NQ/8 blocks of 256 threads.
__global__ __launch_bounds__(256)
void topk_merge_kernel(const float* __restrict__ candS,
                       const int*   __restrict__ candI,
                       const float* __restrict__ pagesf,
                       float* __restrict__ out) {
    const int lane = threadIdx.x & 31;
    const int wave = threadIdx.x >> 5;
    const int q    = blockIdx.x * 8 + wave;

    float t0 = -INFINITY, t1 = -INFINITY;
    int   i0 = 0, i1 = 0;
    MinPos mp = {-INFINITY, lane};

    const float* cs = candS + (size_t)q * (NSPLIT * TOPK);
    const int*   ci = candI + (size_t)q * (NSPLIT * TOPK);
    for (int i = lane; i < NSPLIT * TOPK; i += 32)
        topk_insert32(cs[i], ci[i], lane, t0, t1, i0, i1, mp);

    softmax_gather_out(q, lane, t0, t1, i0, i1, pagesf, out);
}

extern "C" void kernel_launch(void* const* d_in, const int* in_sizes, int n_in,
                              void* d_out, int out_size, void* d_ws, size_t ws_size,
                              hipStream_t stream) {
    (void)in_sizes; (void)n_in; (void)out_size;
    const float* query = (const float*)d_in[0];   // [2, 512, 128] fp32
    const float* pages = (const float*)d_in[1];   // [1, 2048, 64, 128] fp32 -> [131072, 128]
    // d_in[2] is top_k == 64, fixed at compile time.
    float* out = (float*)d_out;                   // [2, 512, 128] fp32

    const size_t poolBytes = (size_t)NROWS * DMEM * sizeof(unsigned short); // 32 MB bf16
    const size_t candElems = (size_t)NQ * NSPLIT * TOPK;                    // 512 K
    const size_t need = poolBytes + candElems * (sizeof(float) + sizeof(int)); // 36 MB

    if (ws_size >= need) {
        __bf16* pagesb = (__bf16*)d_ws;
        float*  candS  = (float*)((char*)d_ws + poolBytes);
        int*    candI  = (int*)((char*)d_ws + poolBytes + candElems * sizeof(float));

        const int cblocks = (NROWS * DMEM) / (256 * 8);    // 8192
        convert_pages_bf16<<<cblocks, 256, 0, stream>>>(pages, pagesb);

        dim3 grid(NQ / QTILE, NSPLIT);                     // 64 x 8 = 512 WGs
        paged_score_kernel<true><<<grid, 256, 0, stream>>>(
            query, pages, pagesb, candS, candI, out);

        topk_merge_kernel<<<NQ / 8, 256, 0, stream>>>(candS, candI, pages, out);
    } else {
        paged_score_kernel<false><<<NQ / QTILE, 256, 0, stream>>>(
            query, pages, nullptr, nullptr, nullptr, out);
    }
}
